// ViSiL_81355270521289
// MI455X (gfx1250) — matrix operand order repaired and checked by reading, not compiled, not run on hardware
//
#include <hip/hip_runtime.h>
#include <hip/hip_bf16.h>

typedef __attribute__((ext_vector_type(16))) __bf16 bf16x16;
typedef __attribute__((ext_vector_type(8)))  __bf16 bf16x8;
typedef __attribute__((ext_vector_type(8)))  float  f32x8;
typedef int vi4 __attribute__((vector_size(16)));   // bare vector int4 for async-LDS builtin

#define NVID 8
#define TT   64
#define RR   9
#define DD   512
#define ROWS_PER_VID (TT*RR)   /* 576 */
#define NPAIR (NVID*NVID)      /* 64  */

#define AS1 __attribute__((address_space(1)))
#define AS3 __attribute__((address_space(3)))

#if defined(__AMDGCN__) && __has_builtin(__builtin_amdgcn_global_load_async_to_lds_b128)
#define HAS_ASYNC 1
#else
#define HAS_ASYNC 0
#endif

// 16B global -> LDS copy: async (ASYNCcnt) on CDNA5, sync fallback otherwise.
__device__ __forceinline__ void cp16_g2l(const __bf16* g, __bf16* l) {
#if HAS_ASYNC
  __builtin_amdgcn_global_load_async_to_lds_b128(
      (AS1 vi4*)(__bf16*)g, (AS3 vi4*)l, 0, 0);
#else
  *(uint4*)l = *(const uint4*)g;
#endif
}
__device__ __forceinline__ void async_wait0() {
#if HAS_ASYNC
  asm volatile("s_wait_asynccnt 0x0" ::: "memory");
#else
  asm volatile("" ::: "memory");
#endif
}
__device__ __forceinline__ void async_wait8() {
#if HAS_ASYNC
  asm volatile("s_wait_asynccnt 0x8" ::: "memory");
#else
  asm volatile("" ::: "memory");
#endif
}

// ---- monotonic f32 <-> u32 encoding so atomicMax(u32) == float max ----
__device__ __forceinline__ unsigned fenc(float f) {
  unsigned u = __float_as_uint(f);
  return (u >> 31) ? ~u : (u | 0x80000000u);
}
__device__ __forceinline__ float fdec(unsigned u) {
  return (u >> 31) ? __uint_as_float(u & 0x7fffffffu) : __uint_as_float(~u);
}

// ---------------- kernel 0: init sim volume + loss slot ----------------
__global__ void visil_init(float* __restrict__ sim, float* __restrict__ out) {
  int i = blockIdx.x * blockDim.x + threadIdx.x;
  if (i < NPAIR * TT * TT) sim[i] = 0.f;
  if (i == 0) out[64] = 0.f;
}

// ---------------- kernel 1: fp32 -> bf16 ----------------
__global__ void visil_cvt(const float* __restrict__ x, __bf16* __restrict__ xb, int n) {
  int i = blockIdx.x * blockDim.x + threadIdx.x;
  if (i < n) xb[i] = (__bf16)x[i];
}

// ---------------- kernel 2: Gram (WMMA bf16) + chamfer ----------------
// grid = (64 pairs, 36 m-tiles), block = 128 (4 waves).
// Wave w owns n-tiles [9w, 9w+9) == chamfer j-groups [16w, 16w+16).
// A strip resident in LDS + register fragments; B streamed via async
// global->LDS DMA, double-buffered per wave, overlapped with WMMA.
__launch_bounds__(128)
__global__ void visil_gram(const __bf16* __restrict__ X, float* __restrict__ sim) {
  const int pair = blockIdx.x;
  const int mt   = blockIdx.y;
  const int a = pair >> 3, bvid = pair & 7;
  const int tid  = threadIdx.x;
  const int wave = tid >> 5, lane = tid & 31;

  __shared__ __align__(16) __bf16 As[16][DD + 16];          // full 16x512 A strip
  __shared__ __align__(16) __bf16 Bs[4][2][16][128 + 16];   // per-wave double-buffered chunk
  __shared__ unsigned Pm[4][16][16];                        // encoded running max P[row][jl]

  // --- Stage A strip (rows a*576 + mt*16 + r) via async DMA: 1024 x 16B ---
  const int aRow0 = a * ROWS_PER_VID + mt * 16;
  for (int e = tid; e < 16 * (DD / 8); e += 128) {
    int r = e >> 6, c8 = e & 63;
    cp16_g2l(&X[(aRow0 + r) * DD + c8 * 8], &As[r][c8 * 8]);
  }
  for (int e = tid; e < 4 * 16 * 16; e += 128) ((unsigned*)Pm)[e] = 0u; // enc-min
  async_wait0();
  __syncthreads();

  const int arow  = lane & 15;              // A row (M) / B column (N) for this lane
  const int kAofs = (lane < 16) ? 0 : 8;    // ISA 16-bit A layout lane-half K offset
  const int kBofs = (lane < 16) ? 0 : 8;    // ISA 16-bit B layout lane-half K offset

  // --- Preload all 16 A K-chunk fragments into registers (128 VGPRs) ---
  bf16x16 afr[16];
  #pragma unroll
  for (int kc = 0; kc < 16; ++kc) {
    bf16x8 alo = *(const bf16x8*)&As[arow][kc * 32 + kAofs];
    bf16x8 ahi = *(const bf16x8*)&As[arow][kc * 32 + kAofs + 16];
    #pragma unroll
    for (int t = 0; t < 8; ++t) { afr[kc][t] = alo[t]; afr[kc][t + 8] = ahi[t]; }
  }

  for (int nt9 = 0; nt9 < 9; ++nt9) {
    const int nt    = wave * 9 + nt9;
    const int bRow0 = bvid * ROWS_PER_VID + nt * 16;
    f32x8 acc = {0.f, 0.f, 0.f, 0.f, 0.f, 0.f, 0.f, 0.f};

    // Prologue: issue chunk 0 (8 async b128 per wave; 16 rows x 128 bf16).
    #pragma unroll
    for (int e8 = 0; e8 < 8; ++e8) {
      int e = lane + e8 * 32, r = e >> 4, c8 = e & 15;
      cp16_g2l(&X[(bRow0 + r) * DD + c8 * 8], &Bs[wave][0][r][c8 * 8]);
    }

    #pragma unroll
    for (int c = 0; c < 4; ++c) {
      if (c < 3) {   // issue chunk c+1 into the other buffer while c is in flight
        #pragma unroll
        for (int e8 = 0; e8 < 8; ++e8) {
          int e = lane + e8 * 32, r = e >> 4, c8 = e & 15;
          cp16_g2l(&X[(bRow0 + r) * DD + (c + 1) * 128 + c8 * 8],
                   &Bs[wave][(c + 1) & 1][r][c8 * 8]);
        }
        async_wait8();     // oldest 8 (= chunk c) complete, chunk c+1 in flight
      } else {
        async_wait0();     // last chunk complete
      }
      const __bf16* bbuf = &Bs[wave][c & 1][arow][0];
      #pragma unroll
      for (int kk = 0; kk < 128; kk += 32) {
        bf16x8 blo = *(const bf16x8*)&bbuf[kk + kBofs];
        bf16x8 bhi = *(const bf16x8*)&bbuf[kk + kBofs + 16];
        bf16x16 bq;
        #pragma unroll
        for (int t = 0; t < 8; ++t) { bq[t] = blo[t]; bq[t + 8] = bhi[t]; }
        acc = __builtin_amdgcn_wmma_f32_16x16x32_bf16(
                  false, afr[c * 4 + (kk >> 5)], false, bq, (short)0, acc, false, false);
      }
    }

    // Chamfer step 1: running max over p within j-groups (wave-local LDS atomics).
    // D layout: this lane holds column N = lane&15, rows M = k + 8*(lane>>4).
    const int cIdx = nt * 16 + (lane & 15);   // within-pair column 0..575
    const int jl   = cIdx / 9 - wave * 16;    // local j-group 0..15 (exact by construction)
    #pragma unroll
    for (int k = 0; k < 8; ++k) {
      int row = k + ((lane >> 4) << 3);
      atomicMax(&Pm[wave][row][jl], fenc(acc[k]));
    }
  }
  __syncthreads();

  // Chamfer step 2: mean over o -> atomicAdd P/9 into sim[pair][i][j].
  for (int e = tid; e < 1024; e += 128) {
    int w = e >> 8, row = (e >> 4) & 15, jl = e & 15;
    float p = fdec(Pm[w][row][jl]) * (1.f / 9.f);
    int i = (mt * 16 + row) / 9;
    int j = w * 16 + jl;
    atomicAdd(&sim[(pair * TT + i) * TT + j], p);
  }
}

// ---------------- kernel 3: conv1(1->32,3x3,same)+ReLU+maxpool2 ----------------
__global__ void visil_conv1pool(const float* __restrict__ sim, const float* __restrict__ w1,
                                const float* __restrict__ b1, float* __restrict__ h1) {
  int idx = blockIdx.x * blockDim.x + threadIdx.x;     // 64*32*32*32
  if (idx >= NPAIR * 32 * 32 * 32) return;
  int x = idx & 31, y = (idx >> 5) & 31, oc = (idx >> 10) & 31, p = idx >> 15;
  float wk[9];
  #pragma unroll
  for (int t = 0; t < 9; ++t) wk[t] = w1[oc * 9 + t];
  float bias = b1[oc];
  const float* img = sim + p * TT * TT;
  float m = -1e30f;
  #pragma unroll
  for (int dy = 0; dy < 2; ++dy)
    #pragma unroll
    for (int dx = 0; dx < 2; ++dx) {
      int iy = 2 * y + dy, ix = 2 * x + dx;
      float acc = bias;
      #pragma unroll
      for (int ky = 0; ky < 3; ++ky) {
        int sy = iy + ky - 1; if (sy < 0 || sy >= TT) continue;
        #pragma unroll
        for (int kx = 0; kx < 3; ++kx) {
          int sx = ix + kx - 1; if (sx < 0 || sx >= TT) continue;
          acc += wk[ky * 3 + kx] * img[sy * TT + sx];
        }
      }
      m = fmaxf(m, fmaxf(acc, 0.f));
    }
  h1[idx] = m;  // ((p*32+oc)*32+y)*32+x
}

// ---------------- kernel 4: conv2(32->64)+ReLU+maxpool2, block = (p,oc) ----------------
__global__ void visil_conv2pool(const float* __restrict__ h1, const float* __restrict__ w2,
                                const float* __restrict__ b2, float* __restrict__ h2) {
  __shared__ float wsm[32 * 9];
  int bx = blockIdx.x;                 // 64*64
  int p = bx >> 6, oc = bx & 63;
  int tid = threadIdx.x;
  for (int e = tid; e < 32 * 9; e += 256) wsm[e] = w2[oc * 32 * 9 + e];
  __syncthreads();
  int x = tid & 15, y = tid >> 4;
  const float* in = h1 + p * 32 * 32 * 32;
  float m = -1e30f;
  for (int dy = 0; dy < 2; ++dy)
    for (int dx = 0; dx < 2; ++dx) {
      int iy = 2 * y + dy, ix = 2 * x + dx;      // 0..31
      float acc = b2[oc];
      for (int ic = 0; ic < 32; ++ic) {
        const float* pl = in + ic * 32 * 32;
        const float* wp = wsm + ic * 9;
        #pragma unroll
        for (int ky = 0; ky < 3; ++ky) {
          int sy = iy + ky - 1; if (sy < 0 || sy >= 32) continue;
          #pragma unroll
          for (int kx = 0; kx < 3; ++kx) {
            int sx = ix + kx - 1; if (sx < 0 || sx >= 32) continue;
            acc += wp[ky * 3 + kx] * pl[sy * 32 + sx];
          }
        }
      }
      m = fmaxf(m, fmaxf(acc, 0.f));
    }
  h2[((p * 64 + oc) * 16 + y) * 16 + x] = m;
}

// ---------------- kernel 5: conv3(64->128)+ReLU, block = (p,oc) ----------------
__global__ void visil_conv3(const float* __restrict__ h2, const float* __restrict__ w3,
                            const float* __restrict__ b3, float* __restrict__ h3) {
  __shared__ float wsm[64 * 9];
  int bx = blockIdx.x;                 // 64*128
  int p = bx >> 7, oc = bx & 127;
  int tid = threadIdx.x;
  for (int e = tid; e < 64 * 9; e += 256) wsm[e] = w3[oc * 64 * 9 + e];
  __syncthreads();
  int x = tid & 15, y = tid >> 4;
  const float* in = h2 + p * 64 * 256;
  float acc = b3[oc];
  for (int ic = 0; ic < 64; ++ic) {
    const float* pl = in + ic * 256;
    const float* wp = wsm + ic * 9;
    #pragma unroll
    for (int ky = 0; ky < 3; ++ky) {
      int sy = y + ky - 1; if (sy < 0 || sy >= 16) continue;
      #pragma unroll
      for (int kx = 0; kx < 3; ++kx) {
        int sx = x + kx - 1; if (sx < 0 || sx >= 16) continue;
        acc += wp[ky * 3 + kx] * pl[sy * 16 + sx];
      }
    }
  }
  h3[((p * 128 + oc) * 16 + y) * 16 + x] = fmaxf(acc, 0.f);
}

// ---------------- kernel 6: conv4(1x1) + loss + chamfer head ----------------
__global__ void visil_head(const float* __restrict__ h3, const float* __restrict__ w4,
                           const float* __restrict__ b4, float* __restrict__ out) {
  __shared__ float red[256];
  __shared__ float rowred[16];
  int p = blockIdx.x;
  int tid = threadIdx.x;               // y*16+x
  const float* in = h3 + p * 128 * 256;
  float h = b4[0];
  for (int c = 0; c < 128; ++c) h += w4[c] * in[c * 256 + tid];
  float loss = fmaxf(-(h + 1.f), 0.f) + fmaxf(h - 1.f, 0.f);
  float s = fminf(fmaxf(h, -1.f), 1.f) * 0.5f + 0.5f;

  red[tid] = s;
  __syncthreads();
  if (tid < 16) {                      // max over width per row
    float m = red[tid * 16];
    for (int k = 1; k < 16; ++k) m = fmaxf(m, red[tid * 16 + k]);
    rowred[tid] = m;
  }
  __syncthreads();
  if (tid == 0) {                      // mean over rows
    float mean = 0.f;
    for (int k = 0; k < 16; ++k) mean += rowred[k];
    out[p] = mean * (1.f / 16.f);
  }
  __syncthreads();
  red[tid] = loss;
  __syncthreads();
  for (int stp = 128; stp > 0; stp >>= 1) {
    if (tid < stp) red[tid] += red[tid + stp];
    __syncthreads();
  }
  if (tid == 0) atomicAdd(&out[64], red[0]);
}

extern "C" void kernel_launch(void* const* d_in, const int* in_sizes, int n_in,
                              void* d_out, int out_size, void* d_ws, size_t ws_size,
                              hipStream_t stream) {
  const float* x  = (const float*)d_in[0];
  const float* w1 = (const float*)d_in[1];
  const float* b1 = (const float*)d_in[2];
  const float* w2 = (const float*)d_in[3];
  const float* b2 = (const float*)d_in[4];
  const float* w3 = (const float*)d_in[5];
  const float* b3 = (const float*)d_in[6];
  const float* w4 = (const float*)d_in[7];
  const float* b4 = (const float*)d_in[8];
  float* out = (float*)d_out;

  char* ws = (char*)d_ws;
  __bf16* Xb = (__bf16*)(ws);                        // 4608*512*2  = 4,718,592 B
  float*  sim = (float*)(ws + 4718592);              // 64*64*64*4  = 1,048,576 B
  float*  h1  = (float*)(ws + 5767168);              // 64*32*32*32*4 = 8,388,608 B
  float*  h2  = (float*)(ws + 14155776);             // 64*64*16*16*4 = 4,194,304 B
  float*  h3  = (float*)(ws + 18350080);             // 64*128*16*16*4 = 8,388,608 B

  const int nX = NVID * TT * RR * DD;                // 2,359,296

  visil_init<<<(NPAIR * TT * TT + 255) / 256, 256, 0, stream>>>(sim, out);
  visil_cvt<<<(nX + 255) / 256, 256, 0, stream>>>(x, Xb, nX);
  visil_gram<<<dim3(NPAIR, 36), 128, 0, stream>>>(Xb, sim);
  visil_conv1pool<<<(NPAIR * 32 * 32 * 32) / 256, 256, 0, stream>>>(sim, w1, b1, h1);
  visil_conv2pool<<<NPAIR * 64, 256, 0, stream>>>(h1, w2, b2, h2);
  visil_conv3<<<NPAIR * 128, 256, 0, stream>>>(h2, w3, b3, h3);
  visil_head<<<NPAIR, 256, 0, stream>>>(h3, w4, b4, out);

  (void)in_sizes; (void)n_in; (void)out_size; (void)ws_size;
}
